// MossAudioTokenizerResidualLFQ_46849503264986
// MI455X (gfx1250) — compile-verified, matrix operand order checked by reading, not run on hardware
//
#include <hip/hip_runtime.h>
#include <hip/hip_bf16.h>

typedef __attribute__((ext_vector_type(16))) __bf16 v16bf;
typedef __attribute__((ext_vector_type(8)))  float  v8f;
typedef __attribute__((ext_vector_type(4)))  unsigned int v4u;
typedef __attribute__((ext_vector_type(4)))  int    v4i;
typedef __attribute__((ext_vector_type(8)))  int    v8i;

#define D_DIM   1024
#define T_DIM   4096
#define B_DIM   8
#define NQ_DIM  32
#define KCODES  1024
#define CD      8
#define TT      16
#define THREADS 256
#define EPSV    1e-12f
#define DPAD    (D_DIM + 8)   // break LDS bank conflicts on stride-D accesses

union BfFrag { v16bf v; __bf16 e[16]; };

// ---- Tensor Data Mover: 2D tile (f32) global -> LDS, per ISA ch.8 D# layout ----
__device__ __forceinline__ void tdm_load_f32_2d(unsigned lds_off, const void* gptr,
                                                unsigned tile_d0, unsigned tile_d1,
                                                unsigned tensor_d0, unsigned tensor_d1,
                                                unsigned stride_d0 /*elements*/) {
    unsigned long long ga = (unsigned long long)gptr;
    v4u g0;
    g0[0] = 1u;                                                  // count=1, user desc
    g0[1] = lds_off;                                             // lds_addr
    g0[2] = (unsigned)ga;                                        // global_addr[31:0]
    g0[3] = (unsigned)((ga >> 32) & 0x01FFFFFFu) | 0x80000000u;  // addr[56:32] | type=2
    v8i g1;
    g1[0] = (int)(2u << 16);                                     // data_size=4B, wg_mask=0
    g1[1] = (int)((tensor_d0 & 0xFFFFu) << 16);                  // tensor_dim0[15:0]
    g1[2] = (int)((tensor_d0 >> 16) | ((tensor_d1 & 0xFFFFu) << 16));
    g1[3] = (int)(((tensor_d1 >> 16) & 0xFFFFu) | (tile_d0 << 16)); // tile_dim0
    g1[4] = (int)(tile_d1 & 0xFFFFu);                            // tile_dim1 (tile_dim2=0)
    g1[5] = (int)stride_d0;                                      // tensor_dim0_stride lo32
    g1[6] = 0;                                                   // stride hi / dim1_stride
    g1[7] = 0;
    v4i z4 = {0, 0, 0, 0};
#if __clang_major__ >= 23
    v8i z8 = {0, 0, 0, 0, 0, 0, 0, 0};
    __builtin_amdgcn_tensor_load_to_lds(g0, g1, z4, z4, z8, 0);
#else
    __builtin_amdgcn_tensor_load_to_lds(g0, g1, z4, z4, 0);
#endif
}

__global__ __launch_bounds__(THREADS)
void rvq_fused_kernel(const float* __restrict__ z,
                      const long long* __restrict__ input_length,
                      const float* __restrict__ W_in,
                      const float* __restrict__ b_in,
                      const float* __restrict__ W_out,
                      const float* __restrict__ b_out,
                      const float* __restrict__ codebooks,
                      float* __restrict__ out)
{
    // Residual master (f32) + masked bf16 mirror, resident across all 32 steps.
    __shared__ float   Rres[D_DIM * TT];          // [d][t]         64 KB (TDM dest)
    __shared__ __bf16  RbfT[TT * DPAD];           // [t][d]         32 KB (padded)
    __shared__ __bf16  winBf[CD * DPAD];          // W_in           16 KB (padded)
    __shared__ __bf16  woutBf[D_DIM * CD];        // W_out          16 KB
    __shared__ __bf16  cbnT[KCODES * CD];         // normalized cb  16 KB
    __shared__ float   boutS[D_DIM];              // TDM dest        4 KB
    __shared__ float   binS[CD];
    __shared__ float   zePart[16 * 16];           // GEMM1 K-split reduction
    __shared__ __bf16  encT[TT * CD];
    __shared__ __bf16  zqT[TT * CD];
    __shared__ unsigned long long bestPack[TT];   // packed (simKey<<32)|invK
    __shared__ long long lenS;

    const int tid      = threadIdx.x;
    const int wave     = tid >> 5;
    const int lane     = tid & 31;
    const int lane15   = lane & 15;
    const int laneHalf = lane >> 4;
    const int bidx     = blockIdx.x / (T_DIM / TT);
    const int tile     = blockIdx.x % (T_DIM / TT);
    const int t0       = tile * TT;

    // ---- TDM: pull the [1024 x 16] f32 residual tile of z straight into LDS ----
    if (wave == 0) {
        const float* ztile = z + (long long)bidx * D_DIM * T_DIM + t0;
        tdm_load_f32_2d((unsigned)(unsigned long long)(void*)Rres, ztile,
                        /*tile_d0=*/TT, /*tile_d1=*/D_DIM,
                        /*tensor_d0=*/T_DIM, /*tensor_d1=*/D_DIM,
                        /*stride_d0=*/T_DIM);
        __builtin_amdgcn_s_wait_tensorcnt(0);
    }
    if (tid == 0) lenS = input_length[bidx];
    __syncthreads();
    const long long lenB = lenS;

    // bf16 mirror := bf16(residual * mask)
    for (int i = tid; i < D_DIM * TT; i += THREADS) {
        int d = i >> 4, t = i & 15;
        float v = Rres[i];
        RbfT[t * DPAD + d] = (__bf16)((t0 + t < lenB) ? v : 0.0f);
    }
    __syncthreads();

    const long long qoutN = (long long)B_DIM * D_DIM * T_DIM;
    const long long idxN  = (long long)NQ_DIM * B_DIM * T_DIM;

    for (int nq = 0; nq < NQ_DIM; ++nq) {
        // ---- async TDM of this step's b_out row (overlaps with staging + GEMM1/2) ----
        if (wave == 0)
            tdm_load_f32_2d((unsigned)(unsigned long long)(void*)boutS,
                            b_out + (long long)nq * D_DIM,
                            /*tile_d0=*/D_DIM, /*tile_d1=*/1,
                            /*tensor_d0=*/D_DIM, /*tensor_d1=*/1,
                            /*stride_d0=*/D_DIM);

        // ---- stage per-step weights (L2-hot: 3 MB total across all steps) ----
        const float* wi = W_in      + (long long)nq * CD * D_DIM;
        const float* wo = W_out     + (long long)nq * D_DIM * CD;
        const float* cb = codebooks + (long long)nq * KCODES * CD;
        for (int i = tid; i < CD * D_DIM; i += THREADS) {
            int c = i >> 10, d = i & 1023;
            winBf[c * DPAD + d] = (__bf16)wi[i];
        }
        for (int i = tid; i < D_DIM * CD; i += THREADS)
            woutBf[i] = (__bf16)wo[i];
        for (int k = tid; k < KCODES; k += THREADS) {
            float c[CD]; float s = 0.f;
            for (int j = 0; j < CD; ++j) { c[j] = cb[k * CD + j]; s += c[j] * c[j]; }
            float inv = 1.0f / fmaxf(sqrtf(s), EPSV);
            for (int j = 0; j < CD; ++j) cbnT[k * CD + j] = (__bf16)(c[j] * inv);
        }
        if (tid < CD)  binS[tid]     = b_in[nq * CD + tid];
        zePart[tid] = 0.f;
        if (tid < TT)  bestPack[tid] = 0ull;
        __syncthreads();

        // ---- GEMM1: z_e(16p x 16t) = W_in(16p x 1024) * mr(1024 x 16t), K split 8 ways ----
        {
            v8f acc = {};
            const int kc0 = wave * 128;
            for (int kk = 0; kk < 4; ++kk) {
                const int kc = kc0 + kk * 32;
                BfFrag a, bm;
                if (lane15 < CD) {
                    int base = lane15 * DPAD + kc + laneHalf * 8;
                    for (int j = 0; j < 8; ++j) a.e[j]     = winBf[base + j];
                    for (int j = 0; j < 8; ++j) a.e[8 + j] = winBf[base + 16 + j];
                } else {
                    for (int j = 0; j < 16; ++j) a.e[j] = (__bf16)0.0f;
                }
                int bb = lane15 * DPAD + kc + laneHalf * 8;
                for (int j = 0; j < 8; ++j) bm.e[j]     = RbfT[bb + j];
                for (int j = 0; j < 8; ++j) bm.e[8 + j] = RbfT[bb + 16 + j];
                acc = __builtin_amdgcn_wmma_f32_16x16x32_bf16(false, a.v, false, bm.v,
                                                              (short)0, acc, false, false);
            }
            for (int r = 0; r < 8; ++r)
                atomicAdd(&zePart[(r + laneHalf * 8) * 16 + lane15], acc[r]);   // ds_add_f32
        }
        __syncthreads();

        // ---- enc = L2-normalize(z_e + b_in) along CD ----
        if (tid < TT) {
            float e[CD]; float s = 0.f;
            for (int c = 0; c < CD; ++c) { e[c] = zePart[c * 16 + tid] + binS[c]; s += e[c] * e[c]; }
            float inv = 1.0f / fmaxf(sqrtf(s), EPSV);
            for (int c = 0; c < CD; ++c) encT[tid * CD + c] = (__bf16)(e[c] * inv);
        }
        __syncthreads();

        // ---- GEMM2: sim(16t x 1024k) tiles; per-lane running argmax, reduce once ----
        {
            BfFrag a;
            if (laneHalf == 0) { for (int j = 0; j < 8; ++j) a.e[j] = encT[lane15 * CD + j]; }
            else               { for (int j = 0; j < 8; ++j) a.e[j] = (__bf16)0.0f; }
            for (int j = 8; j < 16; ++j) a.e[j] = (__bf16)0.0f;

            float bestV[8]; int bestK[8];
            for (int r = 0; r < 8; ++r) { bestV[r] = -3.0e38f; bestK[r] = 0; }

            for (int kt = 0; kt < 8; ++kt) {
                const int k0 = (wave * 8 + kt) * 16;
                BfFrag bm;
                if (laneHalf == 0) { for (int j = 0; j < 8; ++j) bm.e[j] = cbnT[(k0 + lane15) * CD + j]; }
                else               { for (int j = 0; j < 8; ++j) bm.e[j] = (__bf16)0.0f; }
                for (int j = 8; j < 16; ++j) bm.e[j] = (__bf16)0.0f;
                v8f acc = {};
                acc = __builtin_amdgcn_wmma_f32_16x16x32_bf16(false, a.v, false, bm.v,
                                                              (short)0, acc, false, false);
                const int kcol = k0 + lane15;
                for (int r = 0; r < 8; ++r)
                    if (acc[r] > bestV[r]) { bestV[r] = acc[r]; bestK[r] = kcol; }
            }
            // single cross-lane reduction per row (within each 16-lane half)
            for (int r = 0; r < 8; ++r) {
                float val = bestV[r];
                int   kb  = bestK[r];
                for (int off = 1; off < 16; off <<= 1) {
                    float ov = __shfl_xor(val, off, 32);
                    int   ok = __shfl_xor(kb,  off, 32);
                    if (ov > val || (ov == val && ok < kb)) { val = ov; kb = ok; }
                }
                if (lane15 == 0) {
                    unsigned int u = __float_as_uint(val);
                    u = (u & 0x80000000u) ? ~u : (u | 0x80000000u);   // orderable key
                    unsigned long long p =
                        ((unsigned long long)u << 32) | (unsigned int)(KCODES - 1 - kb);
                    atomicMax(&bestPack[r + laneHalf * 8], p);        // ds_max_u64
                }
            }
        }
        // b_out must have landed before GEMM3's epilogue (published by the barrier below)
        if (wave == 0) __builtin_amdgcn_s_wait_tensorcnt(0);
        __syncthreads();

        // ---- select code, emit index, gather raw codebook row (straight-through fwd) ----
        if (tid < TT) {
            unsigned long long p = bestPack[tid];
            int k = KCODES - 1 - (int)(unsigned int)(p & 0xFFFFFFFFull);
            out[qoutN + (((long long)nq * B_DIM + bidx) * T_DIM + t0 + tid)] = (float)k;
            for (int c = 0; c < CD; ++c)
                zqT[tid * CD + c] = (__bf16)cb[k * CD + c];
        }
        __syncthreads();

        // ---- GEMM3: z_q(1024d x 16t) = W_out * cb[idx]; fused residual update ----
        {
            BfFrag bm;
            if (laneHalf == 0) { for (int j = 0; j < 8; ++j) bm.e[j] = zqT[lane15 * CD + j]; }
            else               { for (int j = 0; j < 8; ++j) bm.e[j] = (__bf16)0.0f; }
            for (int j = 8; j < 16; ++j) bm.e[j] = (__bf16)0.0f;
            for (int dt = 0; dt < 8; ++dt) {
                const int d0 = (wave * 8 + dt) * 16;
                BfFrag a;
                if (laneHalf == 0) { for (int j = 0; j < 8; ++j) a.e[j] = woutBf[(d0 + lane15) * CD + j]; }
                else               { for (int j = 0; j < 8; ++j) a.e[j] = (__bf16)0.0f; }
                for (int j = 8; j < 16; ++j) a.e[j] = (__bf16)0.0f;
                v8f acc = {};
                acc = __builtin_amdgcn_wmma_f32_16x16x32_bf16(false, a.v, false, bm.v,
                                                              (short)0, acc, false, false);
                const int dbase = d0 + laneHalf * 8;
                const int t     = lane15;
                const float m   = (t0 + t < lenB) ? 1.0f : 0.0f;
                for (int r = 0; r < 8; ++r) {
                    const int d = dbase + r;
                    float zq = acc[r] + boutS[d];
                    float rn = Rres[d * TT + t] - zq * m;
                    Rres[d * TT + t]   = rn;
                    RbfT[t * DPAD + d] = (__bf16)(rn * m);   // 8 consecutive -> b128 store
                }
            }
        }
        __syncthreads();
    }

    // qout = z - residual_final  (exact, incl. masked positions -> 0)
    for (int i = tid; i < D_DIM * TT; i += THREADS) {
        int d = i >> 4, t = i & 15;
        long long gi = ((long long)bidx * D_DIM + d) * T_DIM + t0 + t;
        out[gi] = z[gi] - Rres[i];
    }
    if (blockIdx.x == 0 && tid < B_DIM)
        out[qoutN + idxN + tid] = (float)input_length[tid];
}

extern "C" void kernel_launch(void* const* d_in, const int* in_sizes, int n_in,
                              void* d_out, int out_size, void* d_ws, size_t ws_size,
                              hipStream_t stream) {
    (void)in_sizes; (void)n_in; (void)out_size; (void)d_ws; (void)ws_size;
    const float*     zP   = (const float*)d_in[0];
    const long long* lenP = (const long long*)d_in[1];
    const float*     wiP  = (const float*)d_in[2];
    const float*     biP  = (const float*)d_in[3];
    const float*     woP  = (const float*)d_in[4];
    const float*     boP  = (const float*)d_in[5];
    const float*     cbP  = (const float*)d_in[6];
    float*           outP = (float*)d_out;

    dim3 grid(B_DIM * (T_DIM / TT));   // 2048 workgroups
    dim3 block(THREADS);               // 8 waves (wave32)
    rvq_fused_kernel<<<grid, block, 0, stream>>>(zP, lenP, wiP, biP, woP, boP, cbP, outP);
}
